// LogSigRNNGenerator_65025804862073
// MI455X (gfx1250) — matrix operand order repaired and checked by reading, not compiled, not run on hardware
//
#include <hip/hip_runtime.h>
#include <hip/hip_bf16.h>

#define DD      16
#define HID     256
#define WIN     64
#define LEN     1000
#define BSZ     512
#define PAIRS   120
#define LSIG    136      // DD + PAIRS
#define KIN     392      // HID + LSIG
#define KINP    416      // padded to multiple of 32
#define CH      111      // 999 = 9 * 111 time-chunk

typedef __attribute__((ext_vector_type(16))) __bf16 v16bf;
typedef __attribute__((ext_vector_type(8)))  __bf16 v8bf;
typedef __attribute__((ext_vector_type(8)))  float  v8f;

// Build a 16-element bf16 fragment from two 16-byte chunks (elems 0..7 / 8..15).
__device__ __forceinline__ v16bf make_frag(const __bf16* p0, const __bf16* p1) {
    v8bf lo = *(const v8bf*)p0;
    v8bf hi = *(const v8bf*)p1;
    v16bf r;
#pragma unroll
    for (int i = 0; i < 8; ++i) { r[i] = lo[i]; r[i + 8] = hi[i]; }
    return r;
}

// ---------------------------------------------------------------------------
// Kernel 0: pack weights to bf16 (W1 padded K 392 -> 416 with zeros).
// Weight rows are [N][K] row-major == B = W^T, exactly what the WMMA B
// fragment wants (two 16B runs of K per lane).
// ---------------------------------------------------------------------------
__global__ void prep_weights(const float* __restrict__ W1, const float* __restrict__ W2,
                             const float* __restrict__ W3, const float* __restrict__ Wl,
                             __bf16* __restrict__ W1b, __bf16* __restrict__ W2b,
                             __bf16* __restrict__ W3b, __bf16* __restrict__ Wlb) {
    int idx = blockIdx.x * blockDim.x + threadIdx.x;
    const int n1  = HID * KINP;   // 106496
    const int nsq = HID * HID;    // 65536
    if (idx < n1) {
        int n = idx / KINP, k = idx - n * KINP;
        W1b[idx] = (k < KIN) ? (__bf16)W1[n * KIN + k] : (__bf16)0.0f;
    } else {
        int q = idx - n1;
        if (q < nsq)            W2b[q]           = (__bf16)W2[q];
        else if (q < 2 * nsq)   W3b[q - nsq]     = (__bf16)W3[q - nsq];
        else if (q < 3 * nsq)   Wlb[q - 2 * nsq] = (__bf16)Wl[q - 2 * nsq];
    }
}

// ---------------------------------------------------------------------------
// Kernel 1: log-signature stream. One block per batch element, 160 threads:
//   tid <120   : one Levy pair (iu,ju) each — running cumsum X_iu, X_ju and
//                running pair-sum C; emit 0.5*(C[t_k] - C[50*floor(t_k/50)])
//   tid 120..135: one dim each — running X_d, emit lvl1 at t_k
// z rows are staged through LDS in chunks so global traffic is exactly 32MB.
// Static schedule: t_k = floor(999*(k+1)/63); s_k = largest multiple of 50 <= t_k.
// Output seq is bf16 [step][b][136]; step 0 zeroed here.
// ---------------------------------------------------------------------------
__global__ void __launch_bounds__(160)
logsig_kernel(const float* __restrict__ z, __bf16* __restrict__ seqB) {
    __shared__ float zrow[CH * DD];
    const int b   = blockIdx.x;
    const int tid = threadIdx.x;
    const float DTS = sqrtf(1.0f / 999.0f);   // all dt equal

    if (tid < LSIG) seqB[(size_t)b * LSIG + tid] = (__bf16)0.0f;  // seq[0] = 0

    int iu = 0, ju = 0;
    if (tid < PAIRS) {
        int p = tid;
        for (int r = 0; r < DD - 1; ++r) {
            int cnt = DD - 1 - r;
            if (p < cnt) { iu = r; ju = r + 1 + p; break; }
            p -= cnt;
        }
    }
    float xi = 0.f, xj = 0.f, c = 0.f, cmark = 0.f;
    int kk = 0;
    int tk = 999 / 63;   // 15: first sample index

    for (int base = 0; base < LEN - 1; base += CH) {
        for (int e = tid; e < CH * DD; e += 160) {
            int i = e >> 4, d = e & 15;
            zrow[e] = z[((size_t)b * LEN + base + 1 + i) * DD + d] * DTS;
        }
        __syncthreads();
        if (tid < PAIRS) {
            for (int i = 0; i < CH; ++i) {
                float dzi = zrow[i * DD + iu];
                float dzj = zrow[i * DD + ju];
                c  += xi * dzj - xj * dzi;   // pair[n] uses X[n] then dX = zs[n+1]
                xi += dzi;  xj += dzj;
                int n1 = base + i + 1;
                if ((n1 % 50) == 0) cmark = c;          // C at multiples of 50 (s_idx)
                if (n1 == tk) {
                    seqB[((size_t)(kk + 1) * BSZ + b) * LSIG + DD + tid] =
                        (__bf16)(0.5f * (c - cmark));
                    ++kk; tk = (999 * (kk + 1)) / 63;
                }
            }
        } else if (tid < PAIRS + DD) {
            int d = tid - PAIRS;
            for (int i = 0; i < CH; ++i) {
                xi += zrow[i * DD + d];                  // X[n+1][d]
                int n1 = base + i + 1;
                if (n1 == tk) {
                    seqB[((size_t)(kk + 1) * BSZ + b) * LSIG + d] = (__bf16)xi;
                    ++kk; tk = (999 * (kk + 1)) / 63;
                }
            }
        }
        __syncthreads();
    }
}

// ---------------------------------------------------------------------------
// Fused GEMM tile: each wave computes 2 N-tiles (32 cols) x 16 rows with
// v_wmma_f32_16x16x32_bf16, A from LDS, B (=W^T rows) from global/L2.
// act: 0=none, 1=relu, 2=tanh.  Writes bf16 to LDS or f32 to global.
// ---------------------------------------------------------------------------
__device__ __forceinline__ void mlp_gemm(
    const __bf16* __restrict__ A, int lda, int ktiles,
    const __bf16* __restrict__ Wg, int ldw,
    const float* __restrict__ bias, int act,
    __bf16* __restrict__ OutLds, float* __restrict__ OutG, int gstride,
    int wave, int lrow, int lhalf) {
    v8f acc[2] = {};
    for (int kk = 0; kk < ktiles; ++kk) {
        int k0 = kk * 32;
        // A fragment (ISA 16-bit A 16x32): lanes<16: K k0..+7 & k0+16..+23,
        // lanes>=16: +8 / +24, row = lane%16.
        const __bf16* pa = A + lrow * lda + k0 + lhalf * 8;
        v16bf a = make_frag(pa, pa + 16);
#pragma unroll
        for (int j = 0; j < 2; ++j) {
            int n = (wave * 2 + j) * 16 + lrow;   // B col = lane%16 within N-tile
            const __bf16* pb = Wg + (size_t)n * ldw + k0 + lhalf * 8;
            v16bf bb = make_frag(pb, pb + 16);    // K +0..7 & +16..23 (+8 for lanes>=16), same as A
            acc[j] = __builtin_amdgcn_wmma_f32_16x16x32_bf16(
                false, a, false, bb, (short)0, acc[j], false, false);
        }
    }
#pragma unroll
    for (int j = 0; j < 2; ++j) {
        int ncol = (wave * 2 + j) * 16 + lrow;
        float bv = bias ? bias[ncol] : 0.0f;
#pragma unroll
        for (int r = 0; r < 8; ++r) {             // D: VGPR r -> row r + 8*lhalf
            float v = acc[j][r] + bv;
            if (act == 1)      v = v > 0.0f ? v : 0.0f;
            else if (act == 2) v = tanhf(v);
            int m = r + lhalf * 8;
            if (OutLds) OutLds[m * HID + ncol] = (__bf16)v;
            else        OutG[(size_t)m * gstride + ncol] = v;
        }
    }
}

// ---------------------------------------------------------------------------
// Kernel 2: fused 64-step RNN. 32 blocks x 256 threads (8 waves). Each block
// owns batch rows [16*bid, 16*bid+16) and the full 256-wide layer, so the
// entire recurrence needs no inter-block traffic: h lives in LDS (as the
// first 256 cols of the layer-1 input buffer), logsig features are copied in
// per step, weights stream from L2/WGP$.
//
// The asm memory clobber each step stops LICM from hoisting the (loop-
// invariant) weight-fragment loads out of the 64-step loop — round 1 showed
// that hoisting spills ~2.4KB/lane to scratch and turns every weight access
// into scratch_load_b128. With the clobber they stay global_load_b128 from
// the ~0.6MB L2-resident weight set.
// ---------------------------------------------------------------------------
__global__ void __launch_bounds__(256)
rnn_kernel(const __bf16* __restrict__ seqB,
           const __bf16* __restrict__ W1b, const __bf16* __restrict__ W2b,
           const __bf16* __restrict__ W3b, const __bf16* __restrict__ Wlb,
           const float* __restrict__ b1, const float* __restrict__ b2,
           const float* __restrict__ b3,
           float* __restrict__ out, unsigned long long updMask) {
    __shared__ __bf16 sIn[16 * KINP];   // [h_last (256) | logsig (136) | pad 0 (24)]
    __shared__ __bf16 sA [16 * HID];
    __shared__ __bf16 sB [16 * HID];
    __shared__ __bf16 sH [16 * HID];

    const int tid   = threadIdx.x;
    const int r0    = blockIdx.x * 16;
    const int wave  = tid >> 5;
    const int lane  = tid & 31;
    const int lrow  = lane & 15;
    const int lhalf = lane >> 4;

    for (int e = tid; e < 16 * KINP; e += 256) sIn[e] = (__bf16)0.0f;  // h0=0, pad=0
    __syncthreads();

#pragma unroll 1
    for (int s = 0; s < WIN; ++s) {
        // Keep per-step memory traffic per-step: no cross-iteration load reuse.
        asm volatile("" ::: "memory");

        // load this step's logsig features into cols 256..391
        for (int e = tid; e < 16 * LSIG; e += 256) {
            int row = e / LSIG, f = e - row * LSIG;
            sIn[row * KINP + HID + f] = seqB[((size_t)s * BSZ + r0 + row) * LSIG + f];
        }
        // speculative prefetch of next step's logsig rows into cache
        if (s + 1 < WIN && tid < 16) {
            __builtin_prefetch(&seqB[((size_t)(s + 1) * BSZ + r0 + tid) * LSIG], 0, 3);
        }
        __syncthreads();

        mlp_gemm(sIn, KINP, KINP / 32, W1b, KINP, b1, 1, sA, nullptr, 0, wave, lrow, lhalf);
        __syncthreads();
        mlp_gemm(sA,  HID,  HID / 32,  W2b, HID,  b2, 1, sB, nullptr, 0, wave, lrow, lhalf);
        __syncthreads();
        mlp_gemm(sB,  HID,  HID / 32,  W3b, HID,  b3, 2, sH, nullptr, 0, wave, lrow, lhalf);
        __syncthreads();
        // y = h @ W_lin^T straight to global out[b][s][:]
        mlp_gemm(sH,  HID,  HID / 32,  Wlb, HID,  nullptr, 0, nullptr,
                 out + (size_t)r0 * WIN * HID + (size_t)s * HID, WIN * HID,
                 wave, lrow, lhalf);

        if ((updMask >> s) & 1ull) {   // last_h = where(upd, h, last_h)
            for (int e = tid; e < 16 * HID; e += 256) {
                int row = e >> 8;
                sIn[row * KINP + (e & 255)] = sH[e];
            }
        }
        __syncthreads();
    }
}

// ---------------------------------------------------------------------------
extern "C" void kernel_launch(void* const* d_in, const int* in_sizes, int n_in,
                              void* d_out, int out_size, void* d_ws, size_t ws_size,
                              hipStream_t stream) {
    const float* z  = (const float*)d_in[0];
    const float* W1 = (const float*)d_in[1];
    const float* b1 = (const float*)d_in[2];
    const float* W2 = (const float*)d_in[3];
    const float* b2 = (const float*)d_in[4];
    const float* W3 = (const float*)d_in[5];
    const float* b3 = (const float*)d_in[6];
    const float* Wl = (const float*)d_in[7];

    char* ws = (char*)d_ws;
    __bf16* W1b  = (__bf16*)(ws);                                  // 256*416*2 = 212992
    __bf16* W2b  = (__bf16*)(ws + 212992);                         // 131072
    __bf16* W3b  = (__bf16*)(ws + 212992 + 131072);
    __bf16* Wlb  = (__bf16*)(ws + 212992 + 2 * 131072);
    __bf16* seqB = (__bf16*)(ws + 212992 + 3 * 131072);            // 64*512*136*2 = 8.9MB

    // upd mask (static schedule): upd[i] for i = ceil(3150*m/999), m=0..19, plus i=63.
    unsigned long long upd = 1ull << 63;
    for (int m = 0; m < 20; ++m) {
        int i = (3150 * m + 998) / 999;
        upd |= (1ull << i);
    }

    const int totalPrep = HID * KINP + 3 * HID * HID;  // 303104
    prep_weights<<<(totalPrep + 255) / 256, 256, 0, stream>>>(W1, W2, W3, Wl,
                                                              W1b, W2b, W3b, Wlb);
    logsig_kernel<<<BSZ, 160, 0, stream>>>(z, seqB);
    rnn_kernel<<<BSZ / 16, 256, 0, stream>>>(seqB, W1b, W2b, W3b, Wlb,
                                             b1, b2, b3, (float*)d_out, upd);
}
